// attentionmodel_27230092656803
// MI455X (gfx1250) — compile-verified, matrix operand order checked by reading, not run on hardware
//
#include <hip/hip_runtime.h>
#include <hip/hip_bf16.h>
#include <math.h>

// ---------------------------------------------------------------------------
// Problem sizes (fixed by the reference)
// ---------------------------------------------------------------------------
#define BB 32
#define LL 1024
#define FF 1024
#define DD 1024

typedef __attribute__((ext_vector_type(16))) __bf16 v16bf;
typedef __attribute__((ext_vector_type(8)))  float  v8f;

union V16 {            // 8 VGPRs worth of bf16 operand
    v16bf v;
    uint4 q[2];
};

__device__ __forceinline__ unsigned short f32_to_bf16(float f) {
    // round-to-nearest-even
    unsigned int u = __float_as_uint(f);
    return (unsigned short)((u + 0x7FFFu + ((u >> 16) & 1u)) >> 16);
}

__device__ __forceinline__ float fast_tanh(float x) {
#if __has_builtin(__builtin_amdgcn_tanhf)
    return __builtin_amdgcn_tanhf(x);      // v_tanh_f32 (gfx1250 TRANS op)
#else
    return tanhf(x);
#endif
}

// ---------------------------------------------------------------------------
// Kernel 1: pack W1 [F=K rows, D cols] fp32 -> bf16 in WMMA B-operand tile
// layout: flat index = ((dt*32 + kt)*32 + lane)*16 + e
//   n    = lane & 15, half = lane >> 4
//   k    = kt*32 + (e&7) + 8*half + ((e>=8) ? 16 : 0)
//   d    = dt*16 + n
// ---------------------------------------------------------------------------
__global__ __launch_bounds__(256) void pack_w1_kernel(
        const float* __restrict__ W1, unsigned short* __restrict__ w1p) {
    int idx  = blockIdx.x * 256 + threadIdx.x;       // 0 .. 2^20-1
    int e    = idx & 15;
    int lane = (idx >> 4) & 31;
    int kt   = (idx >> 9) & 31;
    int dt   = idx >> 14;                             // 0..63
    int half = lane >> 4;
    int n    = lane & 15;
    int k = kt * 32 + (e & 7) + 8 * half + ((e >> 3) << 4);
    int d = dt * 16 + n;
    w1p[idx] = f32_to_bf16(W1[k * DD + d]);
}

// ---------------------------------------------------------------------------
// Kernel 2: h2[b,d] = hidden[b,:] @ W2[:,d] + b2[d] + b1[d]
// (b1 folded in: tanh(h1 + h2) = tanh(x@W1 + b1 + h@W2 + b2))
// ---------------------------------------------------------------------------
__global__ __launch_bounds__(256) void h2_kernel(
        const float* __restrict__ hidden, const float* __restrict__ W2,
        const float* __restrict__ b1,     const float* __restrict__ b2,
        float* __restrict__ h2) {
    int idx = blockIdx.x * 256 + threadIdx.x;        // B*D
    int b = idx >> 10;
    int d = idx & (DD - 1);
    const float* hrow = hidden + b * 1024;
    float acc = b1[d] + b2[d];
    for (int k = 0; k < 1024; ++k)
        acc = fmaf(hrow[k], W2[k * DD + d], acc);
    h2[idx] = acc;
}

// ---------------------------------------------------------------------------
// Kernel 3: fused  score[b,l] = sum_d tanh( (features@W1)[b,l,d] + h2[b,d] ) * Wp[d]
// One block = (batch b, tile of 64 L-rows).  4 waves, each owns 16 D-tiles,
// processed as 8 pairs.  Register tile = 4 L-subtiles x 2 D-tiles
// (8 v8f accumulators): each A fragment feeds 2 WMMAs, each B fragment feeds
// 4 WMMAs  ->  1 ds_load_b128 + 0.5 global_load_b128 per WMMA.
// Per-subtile LDS row pointers are hoisted so all in-loop DS addresses are
// base + immediate (no address VALU -> no WMMA->VALU hazard NOPs).
// ---------------------------------------------------------------------------
#define AT_STRIDE 1032   // 1024 + 8 ushort pad -> row stride 2064B, 16B aligned
#define LROWS 64         // L rows per block (4 subtiles of 16)

#define SMEM_BYTES (LROWS * AT_STRIDE * 2 /*Atile bf16*/ \
                    + DD * 4 /*h2s*/ + DD * 4 /*wps*/ + 4 * LROWS * 4 /*spart*/)

__global__ __launch_bounds__(128) void score_kernel(
        const float* __restrict__ features,          // [B, L, F]
        const float* __restrict__ h2,                // [B, D]
        const float* __restrict__ wp,                // [D]
        const unsigned short* __restrict__ w1p,      // packed bf16 B tiles
        float* __restrict__ score) {                 // [B, L]
    extern __shared__ __align__(16) char smem[];
    unsigned short* Atile = (unsigned short*)smem;                 // [64][AT_STRIDE]
    float* h2s   = (float*)(smem + LROWS * AT_STRIDE * 2);
    float* wps   = h2s + DD;
    float* spart = wps + DD;                                       // [4][64]

    const int b    = blockIdx.x;       // 0..31
    const int ltg  = blockIdx.y;       // 0..15  (group of 64 L rows)
    const int tid  = threadIdx.x;      // 0..127
    const int wave = tid >> 5;
    const int lane = tid & 31;
    const int half = lane >> 4;
    const int nn   = lane & 15;

    // ---- stage A tile (64 x 1024) fp32 -> bf16 into LDS, coalesced ----
    const float* frow = features + ((size_t)b * LL + (size_t)ltg * LROWS) * FF;
    for (int i = tid; i < LROWS * 1024; i += 128) {
        Atile[(i >> 10) * AT_STRIDE + (i & 1023)] = f32_to_bf16(frow[i]);
    }
    for (int i = tid; i < DD; i += 128) {
        h2s[i] = h2[b * DD + i];
        wps[i] = wp[i];
    }
    __syncthreads();

    // ---- per-wave accumulation ----
    float part[4][8];
    #pragma unroll
    for (int s = 0; s < 4; ++s)
        #pragma unroll
        for (int r = 0; r < 8; ++r) part[s][r] = 0.0f;

    // Loop-invariant per-subtile A row pointers (base VGPRs; all in-loop DS
    // accesses are immediate offsets kt*64 / kt*64+32 from these).
    const unsigned short* aRow0 = Atile + (0 * 16 + nn) * AT_STRIDE + 8 * half;
    const unsigned short* aRow1 = Atile + (1 * 16 + nn) * AT_STRIDE + 8 * half;
    const unsigned short* aRow2 = Atile + (2 * 16 + nn) * AT_STRIDE + 8 * half;
    const unsigned short* aRow3 = Atile + (3 * 16 + nn) * AT_STRIDE + 8 * half;

    for (int dtp = 0; dtp < 8; ++dtp) {
        const int dt0 = wave * 16 + dtp * 2;         // D-tile pair (dt0, dt0+1)
        v8f ca[4], cb[4];
        #pragma unroll
        for (int s = 0; s < 4; ++s) {
            ca[s] = (v8f){0.f,0.f,0.f,0.f,0.f,0.f,0.f,0.f};
            cb[s] = (v8f){0.f,0.f,0.f,0.f,0.f,0.f,0.f,0.f};
        }

        const unsigned short* bp0 =
            w1p + (((size_t)dt0 * 32) * 32 + (size_t)lane) * 16;
        const unsigned short* bp1 = bp0 + 32 * 32 * 16;   // next D-tile block

        #pragma unroll 4
        for (int kt = 0; kt < 32; ++kt) {
            // Two B fragments (pre-swizzled, 32B contiguous per lane)
            V16 b0, b1;
            const unsigned short* bk0 = bp0 + (size_t)kt * 512;
            const unsigned short* bk1 = bp1 + (size_t)kt * 512;
            b0.q[0] = *(const uint4*)(bk0); b0.q[1] = *(const uint4*)(bk0 + 8);
            b1.q[0] = *(const uint4*)(bk1); b1.q[1] = *(const uint4*)(bk1 + 8);

            // 4 A fragments (ISA 16-bit A 16x32 layout); each feeds 2 WMMAs
            V16 au;

            au.q[0] = *(const uint4*)(aRow0 + kt * 32);
            au.q[1] = *(const uint4*)(aRow0 + kt * 32 + 16);
            ca[0] = __builtin_amdgcn_wmma_f32_16x16x32_bf16(
                        false, au.v, false, b0.v, (short)0, ca[0], false, false);
            cb[0] = __builtin_amdgcn_wmma_f32_16x16x32_bf16(
                        false, au.v, false, b1.v, (short)0, cb[0], false, false);

            au.q[0] = *(const uint4*)(aRow1 + kt * 32);
            au.q[1] = *(const uint4*)(aRow1 + kt * 32 + 16);
            ca[1] = __builtin_amdgcn_wmma_f32_16x16x32_bf16(
                        false, au.v, false, b0.v, (short)0, ca[1], false, false);
            cb[1] = __builtin_amdgcn_wmma_f32_16x16x32_bf16(
                        false, au.v, false, b1.v, (short)0, cb[1], false, false);

            au.q[0] = *(const uint4*)(aRow2 + kt * 32);
            au.q[1] = *(const uint4*)(aRow2 + kt * 32 + 16);
            ca[2] = __builtin_amdgcn_wmma_f32_16x16x32_bf16(
                        false, au.v, false, b0.v, (short)0, ca[2], false, false);
            cb[2] = __builtin_amdgcn_wmma_f32_16x16x32_bf16(
                        false, au.v, false, b1.v, (short)0, cb[2], false, false);

            au.q[0] = *(const uint4*)(aRow3 + kt * 32);
            au.q[1] = *(const uint4*)(aRow3 + kt * 32 + 16);
            ca[3] = __builtin_amdgcn_wmma_f32_16x16x32_bf16(
                        false, au.v, false, b0.v, (short)0, ca[3], false, false);
            cb[3] = __builtin_amdgcn_wmma_f32_16x16x32_bf16(
                        false, au.v, false, b1.v, (short)0, cb[3], false, false);
        }

        // Epilogue: tanh(+h2) * Wp, accumulate per-row partial score.
        // D layout: lane -> N (=nn within half), VGPR r -> M (lanes>=16: M=r+8)
        const int d0 = dt0 * 16 + nn;
        const int d1 = d0 + 16;
        const float hv0 = h2s[d0], wv0 = wps[d0];
        const float hv1 = h2s[d1], wv1 = wps[d1];
        #pragma unroll
        for (int s = 0; s < 4; ++s)
            #pragma unroll
            for (int r = 0; r < 8; ++r)
                part[s][r] += fast_tanh(ca[s][r] + hv0) * wv0 +
                              fast_tanh(cb[s][r] + hv1) * wv1;
    }

    // ---- reduce over the 16 N-lanes of each half, then across waves ----
    #pragma unroll
    for (int s = 0; s < 4; ++s) {
        #pragma unroll
        for (int r = 0; r < 8; ++r) {
            float v = part[s][r];
            v += __shfl_xor(v, 1);
            v += __shfl_xor(v, 2);
            v += __shfl_xor(v, 4);
            v += __shfl_xor(v, 8);
            if (nn == 0)                     // row m = s*16 + r + 8*half
                spart[wave * LROWS + s * 16 + r + 8 * half] = v;
        }
    }
    __syncthreads();
    if (tid < LROWS) {
        float s = spart[tid] + spart[LROWS + tid] +
                  spart[2 * LROWS + tid] + spart[3 * LROWS + tid];
        score[b * LL + ltg * LROWS + tid] = s;
    }
}

// ---------------------------------------------------------------------------
// Kernel 4: softmax over L per batch row (bp omitted: shift-invariant)
// ---------------------------------------------------------------------------
__global__ __launch_bounds__(256) void softmax_kernel(
        const float* __restrict__ score, float* __restrict__ weights) {
    __shared__ float red[8];
    const int b = blockIdx.x, tid = threadIdx.x;
    const float* s = score + b * LL;

    float m = -INFINITY;
    for (int i = tid; i < LL; i += 256) m = fmaxf(m, s[i]);
    for (int off = 16; off >= 1; off >>= 1) m = fmaxf(m, __shfl_xor(m, off));
    if ((tid & 31) == 0) red[tid >> 5] = m;
    __syncthreads();
    float gmax = fmaxf(fmaxf(fmaxf(red[0], red[1]), fmaxf(red[2], red[3])),
                       fmaxf(fmaxf(red[4], red[5]), fmaxf(red[6], red[7])));
    __syncthreads();

    float sum = 0.0f;
    for (int i = tid; i < LL; i += 256) sum += __expf(s[i] - gmax);
    for (int off = 16; off >= 1; off >>= 1) sum += __shfl_xor(sum, off);
    if ((tid & 31) == 0) red[tid >> 5] = sum;
    __syncthreads();
    float gsum = red[0] + red[1] + red[2] + red[3] +
                 red[4] + red[5] + red[6] + red[7];
    const float inv = 1.0f / gsum;
    for (int i = tid; i < LL; i += 256)
        weights[b * LL + i] = __expf(s[i] - gmax) * inv;
}

// ---------------------------------------------------------------------------
// Kernel 5: context[b,f] = sum_l weights[b,l] * features[b,l,f]
// ---------------------------------------------------------------------------
__global__ __launch_bounds__(256) void context_kernel(
        const float* __restrict__ features, const float* __restrict__ weights,
        float* __restrict__ out) {
    const int b = blockIdx.x;
    const int f = blockIdx.y * 256 + threadIdx.x;
    const float* base = features + ((size_t)b * LL) * FF + f;
    const float* w = weights + b * LL;
    float acc = 0.0f;
    #pragma unroll 4
    for (int l = 0; l < LL; ++l)
        acc = fmaf(w[l], base[(size_t)l * FF], acc);
    out[b * FF + f] = acc;
}

// ---------------------------------------------------------------------------
// Launch
// ---------------------------------------------------------------------------
extern "C" void kernel_launch(void* const* d_in, const int* in_sizes, int n_in,
                              void* d_out, int out_size, void* d_ws, size_t ws_size,
                              hipStream_t stream) {
    const float* features = (const float*)d_in[0];   // [B,L,F]
    const float* hidden   = (const float*)d_in[1];   // [B,H]
    const float* W1       = (const float*)d_in[2];   // [F,D]
    const float* b1       = (const float*)d_in[3];   // [D]
    const float* W2       = (const float*)d_in[4];   // [H,D]
    const float* b2       = (const float*)d_in[5];   // [D]
    const float* Wp       = (const float*)d_in[6];   // [D,1]
    // const float* bp    = (const float*)d_in[7];   // softmax-invariant

    char* ws = (char*)d_ws;
    unsigned short* w1p = (unsigned short*)ws;                      // 2 MB
    float* h2ws      = (float*)(ws + 2u * 1024u * 1024u);           // 128 KB
    float* scorews   = (float*)(ws + 2u * 1024u * 1024u + 128u * 1024u);
    float* weightsws = (float*)(ws + 2u * 1024u * 1024u + 256u * 1024u);
    float* out       = (float*)d_out;                               // [B,F]

    // 1) pack W1 into bf16 WMMA-B layout (1M elements)
    pack_w1_kernel<<<4096, 256, 0, stream>>>(W1, w1p);

    // 2) h2 = hidden@W2 + b2 + b1   (B*D = 32768 threads)
    h2_kernel<<<128, 256, 0, stream>>>(hidden, W2, b1, b2, h2ws);

    // 3) fused big GEMM + tanh + Wp dot -> raw scores
    score_kernel<<<dim3(BB, LL / LROWS), 128, SMEM_BYTES, stream>>>(
        features, h2ws, Wp, w1p, scorews);

    // 4) softmax over L
    softmax_kernel<<<BB, 256, 0, stream>>>(scorews, weightsws);

    // 5) context vector
    context_kernel<<<dim3(BB, FF / 256), 256, 0, stream>>>(
        features, weightsws, out);

    (void)in_sizes; (void)n_in; (void)out_size; (void)ws_size;
}
